// GCN_32169305047251
// MI455X (gfx1250) — compile-verified, matrix operand order checked by reading, not run on hardware
//
#include <hip/hip_runtime.h>
#include <hip/hip_bf16.h>
#include <math.h>

// ---------------------------------------------------------------------------
// Types for CDNA5 WMMA
// ---------------------------------------------------------------------------
typedef __bf16 v16bf __attribute__((ext_vector_type(16)));
typedef __bf16 v8bf  __attribute__((ext_vector_type(8)));
typedef float  v8f   __attribute__((ext_vector_type(8)));

// SiLU with hardware reciprocal (v_rcp_f32) instead of IEEE divide chain.
__device__ __forceinline__ float silu_f(float v) {
    return v * __builtin_amdgcn_rcpf(1.0f + __expf(-v));
}

// ---------------------------------------------------------------------------
// WMMA GEMM on pre-packed bf16 operands.
//   A  : bf16 [M, K]  row-major        (K multiple of 32, zero-padded)
//   BT : bf16 [Nn, K] row-major        (= W transposed, zero-padded)
//   C  : f32  [M, Nn]
// One 16x16 tile per wave. K and FLAGS compile-time => fully unrolled,
// guard-free, branch-free epilogue. Per 32-K step: two 16B loads per
// operand + one v_wmma_f32_16x16x32_bf16.
// 16-bit operand striping per ISA 7.12.2:
//   lane<16 : elems 0..7 -> K=k0+0..7,  elems 8..15 -> K=k0+16..23
//   lane>=16: elems 0..7 -> K=k0+8..15, elems 8..15 -> K=k0+24..31
// D: lane%16 = column, VGPR r -> row = r + 8*(lane>=16)
// FLAGS: bit0 = add bias, bit1 = silu
// ---------------------------------------------------------------------------
template <int K, int FLAGS>
__global__ void wmma_gemm_bf16_kernel(const __bf16* __restrict__ A,
                                      const __bf16* __restrict__ BT,
                                      const float* __restrict__ bias,
                                      float* __restrict__ C,
                                      int M, int Nn) {
    const int wave   = threadIdx.x >> 5;
    const int lane   = threadIdx.x & 31;
    const int tilesN = Nn >> 4;
    const int tile   = blockIdx.x * (blockDim.x >> 5) + wave;
    if (tile >= (M >> 4) * tilesN) return;      // uniform per-wave exit

    const int tm   = tile / tilesN;
    const int tn   = tile % tilesN;
    const int half = lane >> 4;                 // 0 or 1
    const int l16  = lane & 15;

    const v8bf* Ar = (const v8bf*)(A  + (size_t)(tm * 16 + l16) * K);
    const v8bf* Br = (const v8bf*)(BT + (size_t)(tn * 16 + l16) * K);

    v8f acc = {};
#pragma unroll
    for (int k0 = 0; k0 < K; k0 += 32) {
        const int g = (k0 >> 3) + half;         // 8-element group index
        v8bf a0 = Ar[g];
        v8bf a1 = Ar[g + 2];
        v8bf b0 = Br[g];
        v8bf b1 = Br[g + 2];
        v16bf av = __builtin_shufflevector(a0, a1, 0, 1, 2, 3, 4, 5, 6, 7,
                                           8, 9, 10, 11, 12, 13, 14, 15);
        v16bf bv = __builtin_shufflevector(b0, b1, 0, 1, 2, 3, 4, 5, 6, 7,
                                           8, 9, 10, 11, 12, 13, 14, 15);
        acc = __builtin_amdgcn_wmma_f32_16x16x32_bf16(false, av, false, bv,
                                                      (short)0, acc, false, false);
    }

    const int   bcol = tn * 16 + l16;
    const float bv   = (FLAGS & 1) ? bias[bcol] : 0.0f;
#pragma unroll
    for (int r = 0; r < 8; ++r) {
        const int m = tm * 16 + r + half * 8;
        float v = acc[r] + bv;
        if (FLAGS & 2) v = silu_f(v);
        C[(size_t)m * Nn + bcol] = v;
    }
}

// ---------------------------------------------------------------------------
// Operand packing kernels
// ---------------------------------------------------------------------------
// dst[m*Kp + k] = (k < K) ? (bf16)src[m*K + k] : 0
__global__ void cvt_pad_kernel(const float* __restrict__ src, __bf16* __restrict__ dst,
                               int M, int K, int Kp) {
    long long t = (long long)blockIdx.x * blockDim.x + threadIdx.x;
    if (t >= (long long)M * Kp) return;
    const int m = (int)(t / Kp);
    const int k = (int)(t % Kp);
    dst[t] = (k < K) ? (__bf16)src[(size_t)m * K + k] : (__bf16)0.0f;
}

// WT[n*Kp + k] = (k < K) ? (bf16)W[k*Nn + n] : 0
__global__ void transpose_cvt_kernel(const float* __restrict__ W, __bf16* __restrict__ WT,
                                     int K, int Kp, int Nn) {
    long long t = (long long)blockIdx.x * blockDim.x + threadIdx.x;
    if (t >= (long long)Nn * Kp) return;
    const int n = (int)(t / Kp);
    const int k = (int)(t % Kp);
    WT[t] = (k < K) ? (__bf16)W[(size_t)k * Nn + n] : (__bf16)0.0f;
}

// ---------------------------------------------------------------------------
// Graph / elementwise kernels
// ---------------------------------------------------------------------------
__global__ void fill_kernel(float* p, long long n, float v) {
    long long t = (long long)blockIdx.x * blockDim.x + threadIdx.x;
    for (; t < n; t += (long long)gridDim.x * blockDim.x) p[t] = v;
}

__global__ void deg_kernel(const int* __restrict__ col, float* __restrict__ deg, int E) {
    int e = blockIdx.x * blockDim.x + threadIdx.x;
    if (e < E) atomicAdd(&deg[col[e]], 1.0f);
}

__global__ void dinv_kernel(const float* __restrict__ deg, float* __restrict__ dinv, int N) {
    int i = blockIdx.x * blockDim.x + threadIdx.x;
    if (i < N) dinv[i] = rsqrtf(deg[i] + 1.0f);
}

// one lane per 4 features of one edge: gather at src, atomic scatter-add at dst
__global__ void edge_scatter_kernel(const int* __restrict__ row, const int* __restrict__ col,
                                    const float* __restrict__ dinv,
                                    const float* __restrict__ xw,
                                    float* __restrict__ agg, int E, int F) {
    const int lanesPerEdge = F >> 2;
    long long t = (long long)blockIdx.x * blockDim.x + threadIdx.x;
    long long e = t / lanesPerEdge;
    int l = (int)(t % lanesPerEdge);
    if (e >= E) return;
    const int r = row[e];
    const int c = col[e];
    const float dv = dinv[r];
    const float4 v = *(const float4*)(xw + (size_t)r * F + l * 4);
    float* dst = agg + (size_t)c * F + l * 4;
    atomicAdd(dst + 0, dv * v.x);
    atomicAdd(dst + 1, dv * v.y);
    atomicAdd(dst + 2, dv * v.z);
    atomicAdd(dst + 3, dv * v.w);
}

// out = dinv*agg + dinv^2*xw + bias, optional silu
template <int DO_SILU>
__global__ void combine_kernel(const float* __restrict__ dinv,
                               const float* __restrict__ agg,
                               const float* __restrict__ xw,
                               const float* __restrict__ bias,
                               float* __restrict__ out, int N, int F) {
    long long t = (long long)blockIdx.x * blockDim.x + threadIdx.x;
    if (t >= (long long)N * F) return;
    const int i = (int)(t / F);
    const int f = (int)(t % F);
    const float d = dinv[i];
    float v = d * agg[t] + d * d * xw[t] + bias[f];
    if (DO_SILU) v = silu_f(v);
    out[t] = v;
}

// block-per-feature column reduction for BN (biased variance, ddof=0)
__global__ void bn_stats_kernel(const float* __restrict__ z, int M, int F,
                                float* __restrict__ mean, float* __restrict__ var) {
    __shared__ float s_sum[256];
    __shared__ float s_sq[256];
    const int f = blockIdx.x;
    float sum = 0.0f, sq = 0.0f;
    for (int i = threadIdx.x; i < M; i += blockDim.x) {
        float v = z[(size_t)i * F + f];
        sum += v;
        sq  += v * v;
    }
    s_sum[threadIdx.x] = sum;
    s_sq[threadIdx.x]  = sq;
    __syncthreads();
    for (int s = blockDim.x >> 1; s > 0; s >>= 1) {
        if (threadIdx.x < s) {
            s_sum[threadIdx.x] += s_sum[threadIdx.x + s];
            s_sq[threadIdx.x]  += s_sq[threadIdx.x + s];
        }
        __syncthreads();
    }
    if (threadIdx.x == 0) {
        float m = s_sum[0] / (float)M;
        mean[f] = m;
        var[f]  = s_sq[0] / (float)M - m * m;
    }
}

__global__ void bn_apply_silu_kernel(float* __restrict__ z,
                                     const float* __restrict__ mean,
                                     const float* __restrict__ var,
                                     const float* __restrict__ g,
                                     const float* __restrict__ b, int M, int F) {
    long long t = (long long)blockIdx.x * blockDim.x + threadIdx.x;
    if (t >= (long long)M * F) return;
    const int f = (int)(t % F);
    float v = (z[t] - mean[f]) * rsqrtf(var[f] + 1e-5f) * g[f] + b[f];
    z[t] = silu_f(v);
}

// global_add_pool: one lane per 4 features of one node
__global__ void pool_kernel(const float* __restrict__ h, const int* __restrict__ batch,
                            float* __restrict__ hg, int N, int F) {
    const int lanesPerNode = F >> 2;
    long long t = (long long)blockIdx.x * blockDim.x + threadIdx.x;
    long long n = t / lanesPerNode;
    int l = (int)(t % lanesPerNode);
    if (n >= N) return;
    const int g = batch[n];
    const float4 v = *(const float4*)(h + (size_t)n * F + l * 4);
    float* dst = hg + (size_t)g * F + l * 4;
    atomicAdd(dst + 0, v.x);
    atomicAdd(dst + 1, v.y);
    atomicAdd(dst + 2, v.z);
    atomicAdd(dst + 3, v.w);
}

__global__ void concat_kernel(const float* __restrict__ a, const float* __restrict__ b,
                              float* __restrict__ out, int G, int Fa, int Fb) {
    const int Ft = Fa + Fb;
    long long t = (long long)blockIdx.x * blockDim.x + threadIdx.x;
    if (t >= (long long)G * Ft) return;
    const int g = (int)(t / Ft);
    const int f = (int)(t % Ft);
    out[t] = (f < Fa) ? a[(size_t)g * Fa + f] : b[(size_t)g * Fb + (f - Fa)];
}

// final [G,128] @ [128,1] + b
__global__ void final_kernel(const float* __restrict__ t1, const float* __restrict__ W,
                             const float* __restrict__ b, float* __restrict__ out,
                             int G, int F) {
    int g = blockIdx.x * blockDim.x + threadIdx.x;
    if (g >= G) return;
    float s = b[0];
    for (int f = 0; f < F; ++f) s += t1[(size_t)g * F + f] * W[f];
    out[g] = s;
}

// ---------------------------------------------------------------------------
// Host orchestration
// ---------------------------------------------------------------------------
static inline void launch_fill(float* p, long long n, float v, hipStream_t stream) {
    int blocks = (int)((n + 255) / 256);
    if (blocks > 65536) blocks = 65536;
    fill_kernel<<<blocks, 256, 0, stream>>>(p, n, v);
}

// pack A and W to bf16 (K padded to Kp, multiple of 32), then WMMA GEMM.
// Only the (Kp, flags) combinations used by this model are instantiated.
static void gemm_full(const float* A, const float* W, const float* bias, float* C,
                      int M, int K, int Kp, int Nn, int flags,
                      __bf16* Abf, __bf16* WTbf, hipStream_t stream) {
    {
        long long n = (long long)M * Kp;
        cvt_pad_kernel<<<(int)((n + 255) / 256), 256, 0, stream>>>(A, Abf, M, K, Kp);
    }
    {
        long long n = (long long)Nn * Kp;
        transpose_cvt_kernel<<<(int)((n + 255) / 256), 256, 0, stream>>>(W, WTbf, K, Kp, Nn);
    }
    const int tiles  = (M >> 4) * (Nn >> 4);
    const int blocks = (tiles + 7) / 8;   // 8 waves (tiles) per 256-thread block
    if (Kp == 64 && flags == 0) {
        wmma_gemm_bf16_kernel<64, 0><<<blocks, 256, 0, stream>>>(Abf, WTbf, bias, C, M, Nn);
    } else if (Kp == 128 && flags == 0) {
        wmma_gemm_bf16_kernel<128, 0><<<blocks, 256, 0, stream>>>(Abf, WTbf, bias, C, M, Nn);
    } else if (Kp == 128 && flags == 3) {
        wmma_gemm_bf16_kernel<128, 3><<<blocks, 256, 0, stream>>>(Abf, WTbf, bias, C, M, Nn);
    } else if (Kp == 224 && flags == 1) {
        wmma_gemm_bf16_kernel<224, 1><<<blocks, 256, 0, stream>>>(Abf, WTbf, bias, C, M, Nn);
    } else if (Kp == 256 && flags == 3) {
        wmma_gemm_bf16_kernel<256, 3><<<blocks, 256, 0, stream>>>(Abf, WTbf, bias, C, M, Nn);
    }
}

extern "C" void kernel_launch(void* const* d_in, const int* in_sizes, int n_in,
                              void* d_out, int out_size, void* d_ws, size_t ws_size,
                              hipStream_t stream) {
    // ---- inputs (setup_inputs order) ----
    const float* x        = (const float*)d_in[0];   // [N,64]
    const int*   ei       = (const int*)  d_in[1];   // [2,E]
    const int*   batch    = (const int*)  d_in[2];   // [N]
    const float* molf     = (const float*)d_in[3];   // [G,200]
    const float* gcW0     = (const float*)d_in[4];
    const float* gcb0     = (const float*)d_in[5];
    const float* gcW1     = (const float*)d_in[6];
    const float* gcb1     = (const float*)d_in[7];
    const float* gcW2     = (const float*)d_in[8];
    const float* gcb2     = (const float*)d_in[9];
    const float* bn_gc_g  = (const float*)d_in[10];
    const float* bn_gc_b  = (const float*)d_in[11];
    const float* fcmW0    = (const float*)d_in[12];
    const float* fcmb0    = (const float*)d_in[13];
    const float* fcmW1    = (const float*)d_in[14];
    const float* fcmb1    = (const float*)d_in[15];
    const float* bn_m_g   = (const float*)d_in[16];
    const float* bn_m_b   = (const float*)d_in[17];
    const float* fcW0     = (const float*)d_in[18];
    const float* fcb0     = (const float*)d_in[19];
    const float* fcW1     = (const float*)d_in[20];
    const float* fcb1     = (const float*)d_in[21];
    const float* fcW2     = (const float*)d_in[22];
    const float* fcb2     = (const float*)d_in[23];
    float* out = (float*)d_out;

    const int N = in_sizes[2];
    const int E = in_sizes[1] / 2;
    const int F_NODE = in_sizes[0] / N;        // 64
    const int F_MOL  = 200;
    const int F_MOLP = 224;                    // padded to multiple of 32
    const int G = in_sizes[3] / F_MOL;         // 2048
    const int D0 = 128, D1 = 128, D2 = 256, D3 = 128;

    const int* row = ei;
    const int* col = ei + E;

    // ---- workspace layout ----
    float* ws = (float*)d_ws;
    size_t off = 0;
    float* deg   = ws + off; off += (size_t)N;
    float* dinv  = ws + off; off += (size_t)N;
    float* xw    = ws + off; off += (size_t)N * D0;
    float* agg   = ws + off; off += (size_t)N * D0;
    float* h0    = ws + off; off += (size_t)N * D0;
    float* h1    = ws + off; off += (size_t)N * D0;
    float* hg    = ws + off; off += (size_t)G * D1;
    float* hm0   = ws + off; off += (size_t)G * D0;
    float* hm1   = ws + off; off += (size_t)G * D1;
    float* hcat  = ws + off; off += (size_t)G * 2 * D1;
    float* t0    = ws + off; off += (size_t)G * D2;
    float* t1    = ws + off; off += (size_t)G * D3;
    float* mean0 = ws + off; off += 256;
    float* var0  = ws + off; off += 256;
    off = (off + 7) & ~(size_t)7;              // 32B-align the bf16 region
    __bf16* Abf  = (__bf16*)(ws + off); off += ((size_t)N * D0 + 1) / 2;  // bf16[M*Kmax]
    __bf16* WTbf = (__bf16*)(ws + off); off += ((size_t)256 * 256 + 1) / 2;
    (void)ws_size; (void)n_in; (void)out_size;

    const long long NF = (long long)N * D0;
    const int blkNF   = (int)((NF + 255) / 256);
    const int blkEdge = (int)(((long long)E * (D0 / 4) + 255) / 256);
    const int blkPool = (int)(((long long)N * (D1 / 4) + 255) / 256);

    // ---- degree / dinv ----
    launch_fill(deg, N, 0.0f, stream);
    deg_kernel<<<(E + 255) / 256, 256, 0, stream>>>(col, deg, E);
    dinv_kernel<<<(N + 255) / 256, 256, 0, stream>>>(deg, dinv, N);

    // ---- GCN layer 0: x[N,64] -> h0[N,128], BN + SiLU ----
    gemm_full(x, gcW0, nullptr, xw, N, F_NODE, F_NODE, D0, 0, Abf, WTbf, stream);
    launch_fill(agg, NF, 0.0f, stream);
    edge_scatter_kernel<<<blkEdge, 256, 0, stream>>>(row, col, dinv, xw, agg, E, D0);
    combine_kernel<0><<<blkNF, 256, 0, stream>>>(dinv, agg, xw, gcb0, h0, N, D0);
    bn_stats_kernel<<<D0, 256, 0, stream>>>(h0, N, D0, mean0, var0);
    bn_apply_silu_kernel<<<blkNF, 256, 0, stream>>>(h0, mean0, var0, bn_gc_g, bn_gc_b, N, D0);

    // ---- GCN layer 1: h0 -> h1, SiLU ----
    gemm_full(h0, gcW1, nullptr, xw, N, D0, D0, D0, 0, Abf, WTbf, stream);
    launch_fill(agg, NF, 0.0f, stream);
    edge_scatter_kernel<<<blkEdge, 256, 0, stream>>>(row, col, dinv, xw, agg, E, D0);
    combine_kernel<1><<<blkNF, 256, 0, stream>>>(dinv, agg, xw, gcb1, h1, N, D0);

    // ---- GCN layer 2: h1 -> h0, SiLU ----
    gemm_full(h1, gcW2, nullptr, xw, N, D0, D0, D1, 0, Abf, WTbf, stream);
    launch_fill(agg, NF, 0.0f, stream);
    edge_scatter_kernel<<<blkEdge, 256, 0, stream>>>(row, col, dinv, xw, agg, E, D1);
    combine_kernel<1><<<blkNF, 256, 0, stream>>>(dinv, agg, xw, gcb2, h0, N, D1);

    // ---- global add pool: h0 -> hg[G,128] ----
    launch_fill(hg, (long long)G * D1, 0.0f, stream);
    pool_kernel<<<blkPool, 256, 0, stream>>>(h0, batch, hg, N, D1);

    // ---- molecular branch ----
    gemm_full(molf, fcmW0, fcmb0, hm0, G, F_MOL, F_MOLP, D0, /*bias*/ 1, Abf, WTbf, stream);
    bn_stats_kernel<<<D0, 256, 0, stream>>>(hm0, G, D0, mean0, var0);
    bn_apply_silu_kernel<<<(int)(((long long)G * D0 + 255) / 256), 256, 0, stream>>>(
        hm0, mean0, var0, bn_m_g, bn_m_b, G, D0);
    gemm_full(hm0, fcmW1, fcmb1, hm1, G, D0, D0, D1, /*bias|silu*/ 3, Abf, WTbf, stream);

    // ---- head ----
    concat_kernel<<<(int)(((long long)G * 2 * D1 + 255) / 256), 256, 0, stream>>>(
        hg, hm1, hcat, G, D1, D1);
    gemm_full(hcat, fcW0, fcb0, t0, G, 2 * D1, 2 * D1, D2, 3, Abf, WTbf, stream);
    gemm_full(t0, fcW1, fcb1, t1, G, D2, D2, D3, 3, Abf, WTbf, stream);
    final_kernel<<<(G + 255) / 256, 256, 0, stream>>>(t1, fcW2, fcb2, out, G, D3);
}
